// DeformableConvNet_57982058496542
// MI455X (gfx1250) — compile-verified, matrix operand order checked by reading, not run on hardware
//
#include <hip/hip_runtime.h>
#include <hip/hip_bf16.h>

typedef __attribute__((ext_vector_type(16))) _Float16 v16h;
typedef __attribute__((ext_vector_type(8)))  _Float16 h8;
typedef __attribute__((ext_vector_type(8)))  float    v8f;

// Problem constants (from reference setup_inputs)
constexpr int CH = 64;    // input channels C
constexpr int OC = 64;    // output channels O
constexpr int HH = 128;
constexpr int WW = 128;
constexpr int NKK = 9;    // 3x3 taps
constexpr int LDA = 72;   // padded LDS row in halfs: 144B = 9*16B -> aligned b128, conflict-free

// Load a v16h WMMA fragment from an LDS tile with row stride LDA.
// Per the CDNA5 ISA 16-bit A/B 16x32 layout: lane&15 = row (M or N),
// sel = lane>>4; elements 0..7 -> K = e0..e0+7, elements 8..15 -> K = e0+16..e0+23
// where e0 = kchunk*32 + 8*sel. Two contiguous 16B loads per fragment.
__device__ inline v16h load_frag(const _Float16* __restrict__ base, int row, int e0) {
  const h8* p0 = (const h8*)(base + row * LDA + e0);
  const h8* p1 = (const h8*)(base + row * LDA + e0 + 16);
  h8 lo = *p0;
  h8 hi = *p1;
  v16h r;
#pragma unroll
  for (int i = 0; i < 8; ++i) { r[i] = lo[i]; r[i + 8] = hi[i]; }
  return r;
}

// ---------------------------------------------------------------------------
// Kernel 1: offset conv. off[n][kk*2+s][h][w] = conv3x3(x, w_off) + b_off
// GEMM: M=128 pixels (one row), N=18 (padded to 32), K=576 ordered (kp, c).
// ---------------------------------------------------------------------------
__global__ __launch_bounds__(256) void k_offsets(const float* __restrict__ x,
                                                 const float* __restrict__ w_off,
                                                 const float* __restrict__ b_off,
                                                 float* __restrict__ off_ws) {
  extern __shared__ char dsm[];
  _Float16* Bt = (_Float16*)dsm;                                  // [9][32][LDA]
  _Float16* A  = (_Float16*)(dsm + (size_t)9 * 32 * LDA * 2);     // [128][LDA]

  const int wg = blockIdx.x;          // n*H + h
  const int n  = wg / HH;
  const int h  = wg % HH;
  const int tid = threadIdx.x;

  // Stage B transposed: Bt[kp][oc][c] = w_off[oc][c][kp] (oc >= 18 padded 0)
  for (int i = tid; i < 9 * 32 * 64; i += 256) {
    int c  = i & 63;
    int oc = (i >> 6) & 31;
    int kp = i >> 11;
    float v = (oc < 18) ? w_off[(oc * CH + c) * NKK + kp] : 0.0f;
    Bt[(kp * 32 + oc) * LDA + c] = (_Float16)v;
  }

  const int lane = tid & 31, wv = tid >> 5;
  const int row = lane & 15, sel = lane >> 4;
  v8f acc0 = {};
  v8f acc1 = {};

  const int p = tid & 127;            // pixel (w) this thread stages
  const int chalf = tid >> 7;         // 0/1 -> channels [0,32) / [32,64)
  const float* xn = x + (size_t)n * CH * HH * WW;

  for (int kp = 0; kp < 9; ++kp) {
    const int ki = kp / 3, kj = kp % 3;
    const int yy = h + ki - 1;
    const int xx = p + kj - 1;
    const bool vld = (yy >= 0) && (yy < HH) && (xx >= 0) && (xx < WW);
    __syncthreads();
    if (vld) {
      const float* src = xn + (size_t)(chalf * 32) * HH * WW + yy * WW + xx;
#pragma unroll 4
      for (int i = 0; i < 32; ++i)
        A[p * LDA + chalf * 32 + i] = (_Float16)src[(size_t)i * HH * WW];
    } else {
#pragma unroll 4
      for (int i = 0; i < 32; ++i)
        A[p * LDA + chalf * 32 + i] = (_Float16)0.0f;
    }
    __syncthreads();

    const _Float16* Bk = Bt + (size_t)kp * 32 * LDA;
#pragma unroll
    for (int kc = 0; kc < 2; ++kc) {
      const int e0 = kc * 32 + 8 * sel;
      v16h a  = load_frag(A, wv * 16 + row, e0);
      v16h b0 = load_frag(Bk, row, e0);
      v16h b1 = load_frag(Bk, 16 + row, e0);
      acc0 = __builtin_amdgcn_wmma_f32_16x16x32_f16(false, a, false, b0, (short)0, acc0, false, false);
      acc1 = __builtin_amdgcn_wmma_f32_16x16x32_f16(false, a, false, b1, (short)0, acc1, false, false);
    }
  }

  // Store: C/D layout -> VGPR r: M = r + 8*sel, N = lane&15
  const int oc_l = lane & 15;
  const int wbase = wv * 16 + 8 * sel;
  {
    const float bb = b_off[oc_l];
    float* op = off_ws + (((size_t)n * 18 + oc_l) * HH + h) * WW;
#pragma unroll
    for (int r = 0; r < 8; ++r) op[wbase + r] = acc0[r] + bb;
  }
  if (oc_l < 2) {
    const int oc = 16 + oc_l;
    const float bb = b_off[oc];
    float* op = off_ws + (((size_t)n * 18 + oc) * HH + h) * WW;
#pragma unroll
    for (int r = 0; r < 8; ++r) op[wbase + r] = acc1[r] + bb;
  }
}

// ---------------------------------------------------------------------------
// Kernel 2: fused bilinear sampling + output GEMM.
// Per workgroup: one (n,h) row. M=128 px, N=64 oc, K=576 ordered (kp, c).
// A tile built per kp in LDS via 4-corner masked bilinear gather (L2-resident x),
// avoiding the 151 MB 'sampled' intermediate entirely.
// ---------------------------------------------------------------------------
__global__ __launch_bounds__(256) void k_main(const float* __restrict__ x,
                                              const float* __restrict__ w_def,
                                              const float* __restrict__ b_def,
                                              const float* __restrict__ off_ws,
                                              float* __restrict__ out) {
  extern __shared__ char dsm[];
  _Float16* Bt = (_Float16*)dsm;                                  // [9][64][LDA]
  _Float16* A  = (_Float16*)(dsm + (size_t)9 * 64 * LDA * 2);     // [128][LDA]

  const int wg = blockIdx.x;
  const int n  = wg / HH;
  const int h  = wg % HH;
  const int tid = threadIdx.x;

  // Stage B transposed: Bt[kp][oc][c] = w_def[oc][c][kp]
  for (int i = tid; i < 9 * 64 * 64; i += 256) {
    int c  = i & 63;
    int oc = (i >> 6) & 63;
    int kp = i >> 12;
    Bt[(kp * 64 + oc) * LDA + c] = (_Float16)w_def[(oc * CH + c) * NKK + kp];
  }

  const int lane = tid & 31, wv = tid >> 5;
  const int row = lane & 15, sel = lane >> 4;
  v8f acc[4] = {};

  const int p = tid & 127;
  const int chalf = tid >> 7;
  const float* xn   = x + (size_t)n * CH * HH * WW;
  const float* offn = off_ws + (size_t)n * 18 * HH * WW;

  for (int kp = 0; kp < 9; ++kp) {
    const int ki = kp / 3, kj = kp % 3;
    const float dy = offn[((2 * kp + 0) * HH + h) * WW + p];
    const float dx = offn[((2 * kp + 1) * HH + h) * WW + p];
    const float py = (float)(h + ki - 1) + dy;
    const float px = (float)(p + kj - 1) + dx;
    const float y0f = floorf(py);
    const float x0f = floorf(px);
    const int y0 = (int)y0f, x0 = (int)x0f;
    const float wy = py - y0f, wx = px - x0f;

    const int y0c = min(max(y0, 0), HH - 1), y1c = min(max(y0 + 1, 0), HH - 1);
    const int x0c = min(max(x0, 0), WW - 1), x1c = min(max(x0 + 1, 0), WW - 1);
    const bool vy0 = (y0 >= 0) && (y0 < HH),   vy1 = (y0 + 1 >= 0) && (y0 + 1 < HH);
    const bool vx0 = (x0 >= 0) && (x0 < WW),   vx1 = (x0 + 1 >= 0) && (x0 + 1 < WW);
    const float w00 = (1.0f - wy) * (1.0f - wx) * ((vy0 && vx0) ? 1.0f : 0.0f);
    const float w01 = (1.0f - wy) * wx          * ((vy0 && vx1) ? 1.0f : 0.0f);
    const float w10 = wy * (1.0f - wx)          * ((vy1 && vx0) ? 1.0f : 0.0f);
    const float w11 = wy * wx                   * ((vy1 && vx1) ? 1.0f : 0.0f);
    const int i00 = y0c * WW + x0c, i01 = y0c * WW + x1c;
    const int i10 = y1c * WW + x0c, i11 = y1c * WW + x1c;

    __syncthreads();
    {
      const float* xc = xn + (size_t)(chalf * 32) * HH * WW;
#pragma unroll 4
      for (int i = 0; i < 32; ++i) {
        const float* b = xc + (size_t)i * HH * WW;
        float s = w00 * b[i00] + w01 * b[i01] + w10 * b[i10] + w11 * b[i11];
        A[p * LDA + chalf * 32 + i] = (_Float16)s;
      }
    }
    __syncthreads();

    const _Float16* Bk = Bt + (size_t)kp * 64 * LDA;
#pragma unroll
    for (int kc = 0; kc < 2; ++kc) {
      const int e0 = kc * 32 + 8 * sel;
      v16h a = load_frag(A, wv * 16 + row, e0);
#pragma unroll
      for (int nt = 0; nt < 4; ++nt) {
        v16h b = load_frag(Bk, nt * 16 + row, e0);
        acc[nt] = __builtin_amdgcn_wmma_f32_16x16x32_f16(false, a, false, b, (short)0, acc[nt], false, false);
      }
    }
  }

  // Store D + bias. M = r + 8*sel (pixel within wave tile), N = lane&15 (oc in N-tile).
  const int oc_l = lane & 15;
  const int wbase = wv * 16 + 8 * sel;
#pragma unroll
  for (int nt = 0; nt < 4; ++nt) {
    const int oc = nt * 16 + oc_l;
    const float bb = b_def[oc];
    float* op = out + (((size_t)n * OC + oc) * HH + h) * WW;
#pragma unroll
    for (int r = 0; r < 8; ++r) op[wbase + r] = acc[nt][r] + bb;
  }
}

extern "C" void kernel_launch(void* const* d_in, const int* in_sizes, int n_in,
                              void* d_out, int out_size, void* d_ws, size_t ws_size,
                              hipStream_t stream) {
  const float* x     = (const float*)d_in[0];
  const float* w_off = (const float*)d_in[1];
  const float* b_off = (const float*)d_in[2];
  const float* w_def = (const float*)d_in[3];
  const float* b_def = (const float*)d_in[4];
  float* out = (float*)d_out;
  float* off_ws = (float*)d_ws;   // [N][18][H][W] fp32 offsets scratch (4.7 MB)

  const int NB = in_sizes[0] / (CH * HH * WW);   // batch (=4)
  const int grid = NB * HH;                      // one (n,h) row per workgroup

  const size_t sm1 = (size_t)(9 * 32 * LDA + 128 * LDA) * sizeof(_Float16); // ~60 KB
  const size_t sm2 = (size_t)(9 * 64 * LDA + 128 * LDA) * sizeof(_Float16); // ~99 KB

  k_offsets<<<grid, 256, sm1, stream>>>(x, w_off, b_off, off_ws);
  k_main<<<grid, 256, sm2, stream>>>(x, w_def, b_def, off_ws, out);
}
